// SCNN_gen_71485435675282
// MI455X (gfx1250) — matrix operand order repaired and checked by reading, not compiled, not run on hardware
//
#include <hip/hip_runtime.h>
#include <hip/hip_bf16.h>

typedef __bf16 bf16_t;
typedef __attribute__((ext_vector_type(16))) __bf16 v16bf;
typedef __attribute__((ext_vector_type(8)))  __bf16 v8bf;
typedef __attribute__((ext_vector_type(8)))  float  v8f;
typedef __attribute__((ext_vector_type(4)))  float  v4f;

#define B_DIM    32
#define S_DIM    8
#define BS_DIM   256      // B*S rows through the conv stacks
#define IN_DIM   8192     // N0+N1+N2
#define H_DIM    1024
#define OUT_DIM  256
#define NFILT    4

// ---------------------------------------------------------------------------
// Elementwise prep kernels
// ---------------------------------------------------------------------------

// xs [BS, n] f32  ->  X [BS, IN_DIM] bf16 at column offset colOff
__global__ void cvt_x_kernel(const float* __restrict__ src, bf16_t* __restrict__ dst,
                             int n, int colOff, int total) {
  int t = blockIdx.x * blockDim.x + threadIdx.x;
  if (t >= total) return;
  int row = t / n, col = t - row * n;
  dst[(size_t)row * IN_DIM + colOff + col] = (bf16_t)src[t];
}

// W [K, N] f32 row-major  ->  WT [N, K] bf16 (so B fragments read contiguous K)
__global__ void transpose_cvt_kernel(const float* __restrict__ W, bf16_t* __restrict__ WT,
                                     int K, int N, int total) {
  int t = blockIdx.x * blockDim.x + threadIdx.x;
  if (t >= total) return;
  int j = t / K, k = t - j * K;
  WT[t] = (bf16_t)W[(size_t)k * N + j];
}

// Hcat [BS, IN_DIM] f32 -> sum over S -> Hb [B, IN_DIM] bf16
__global__ void seqsum_kernel(const float* __restrict__ Hcat, bf16_t* __restrict__ Hb,
                              int total) {
  int t = blockIdx.x * blockDim.x + threadIdx.x;
  if (t >= total) return;
  int b = t / IN_DIM, c = t - b * IN_DIM;
  float s = 0.f;
#pragma unroll
  for (int q = 0; q < S_DIM; ++q)
    s += Hcat[(size_t)(b * S_DIM + q) * IN_DIM + c];
  Hb[t] = (bf16_t)s;
}

// ---------------------------------------------------------------------------
// Conv-stack GEMM:  out[r, m] = act( sum_k X[r, k] * L[m, k] )
// One wave per block. Wave owns one 16-col N-tile and 8 M-tiles (128 rows).
// B fragment = 16 contiguous fp32 of an L row, converted to bf16 in-register.
// Epilogue fuses  sum_f relu(val * Wc[f, m]).
// ---------------------------------------------------------------------------
__global__ void __launch_bounds__(32)
conv_gemm_kernel(const bf16_t* __restrict__ X,    // [BS, IN_DIM] bf16 (+catOff cols)
                 const float*  __restrict__ L,    // [n, n] f32
                 const float*  __restrict__ WcL,  // [NFILT, n] f32 (layer slice)
                 bf16_t* __restrict__ Xn,         // [BS, IN_DIM] bf16 or null
                 float*  __restrict__ Hcat,       // [BS, IN_DIM] f32 or null
                 int n, int catOff) {
  const int lane = threadIdx.x;
  const int nloc = lane & 15;
  const int half = lane >> 4;
  const int nBase = blockIdx.x * 16;
  const int mBase = blockIdx.y * 128;
  const int colG  = nBase + nloc;

  v8f acc[8];
#pragma unroll
  for (int i = 0; i < 8; ++i) acc[i] = (v8f){0.f,0.f,0.f,0.f,0.f,0.f,0.f,0.f};

  const float* lrow = L + (size_t)colG * n + half * 16;

  for (int k = 0; k < n; k += 32) {
    // ---- B fragment: B[kk, c] = L[nBase+c, k+kk]; 16 contiguous fp32 -> bf16
    v4f f0 = *(const v4f*)(lrow + k);
    v4f f1 = *(const v4f*)(lrow + k + 4);
    v4f f2 = *(const v4f*)(lrow + k + 8);
    v4f f3 = *(const v4f*)(lrow + k + 12);
    v16bf bfrag;
#pragma unroll
    for (int i = 0; i < 4; ++i) {
      bfrag[i]      = (__bf16)f0[i];
      bfrag[4 + i]  = (__bf16)f1[i];
      bfrag[8 + i]  = (__bf16)f2[i];
      bfrag[12 + i] = (__bf16)f3[i];
    }
    if (k + 32 < n) __builtin_prefetch((const void*)(lrow + k + 32), 0, 0);

#pragma unroll
    for (int mt = 0; mt < 8; ++mt) {
      int row = mBase + mt * 16 + nloc;
      const bf16_t* xr = X + (size_t)row * IN_DIM + catOff + k + half * 16;
      v8bf alo = *(const v8bf*)(xr);
      v8bf ahi = *(const v8bf*)(xr + 8);
      v16bf afrag = __builtin_shufflevector(alo, ahi,
          0, 1, 2, 3, 4, 5, 6, 7, 8, 9, 10, 11, 12, 13, 14, 15);
      acc[mt] = __builtin_amdgcn_wmma_f32_16x16x32_bf16(
          false, afrag, false, bfrag, (short)0, acc[mt], false, false);
    }
  }

  // ---- fused filter activation: sum_f relu(val * Wc[f, col])
  float w0 = WcL[0 * n + colG];
  float w1 = WcL[1 * n + colG];
  float w2 = WcL[2 * n + colG];
  float w3 = WcL[3 * n + colG];

#pragma unroll
  for (int mt = 0; mt < 8; ++mt) {
#pragma unroll
    for (int v = 0; v < 8; ++v) {
      int rowG = mBase + mt * 16 + half * 8 + v;
      float t = acc[mt][v];
      float r = fmaxf(t * w0, 0.f) + fmaxf(t * w1, 0.f) +
                fmaxf(t * w2, 0.f) + fmaxf(t * w3, 0.f);
      size_t idx = (size_t)rowG * IN_DIM + catOff + colG;
      if (Xn)   Xn[idx]   = (bf16_t)r;
      if (Hcat) Hcat[idx] = r;
    }
  }
}

// ---------------------------------------------------------------------------
// MLP GEMM:  out[r, j] = act( sum_k A[r, k] * WT[j, k] + bias[j] )
// A bf16 [M, K] compact, WT bf16 [N, K] (pre-transposed weights).
// ---------------------------------------------------------------------------
__global__ void __launch_bounds__(32)
mlp_gemm_kernel(const bf16_t* __restrict__ A,
                const bf16_t* __restrict__ WT,
                const float*  __restrict__ bias,
                bf16_t* __restrict__ outBf,   // [M, N] or null
                float*  __restrict__ outF,    // [M, N] or null
                int K, int N, int doRelu) {
  const int lane = threadIdx.x;
  const int nloc = lane & 15;
  const int half = lane >> 4;
  const int nBase = blockIdx.x * 16;
  const int mBase = blockIdx.y * 16;
  const int colG  = nBase + nloc;

  v8f acc = (v8f){0.f,0.f,0.f,0.f,0.f,0.f,0.f,0.f};

  const bf16_t* wrow = WT + (size_t)colG * K + half * 16;
  const bf16_t* arow = A + (size_t)(mBase + nloc) * K + half * 16;

  for (int k = 0; k < K; k += 32) {
    v8bf blo = *(const v8bf*)(wrow + k);
    v8bf bhi = *(const v8bf*)(wrow + k + 8);
    v16bf bfrag = __builtin_shufflevector(blo, bhi,
        0, 1, 2, 3, 4, 5, 6, 7, 8, 9, 10, 11, 12, 13, 14, 15);
    v8bf alo = *(const v8bf*)(arow + k);
    v8bf ahi = *(const v8bf*)(arow + k + 8);
    v16bf afrag = __builtin_shufflevector(alo, ahi,
        0, 1, 2, 3, 4, 5, 6, 7, 8, 9, 10, 11, 12, 13, 14, 15);
    acc = __builtin_amdgcn_wmma_f32_16x16x32_bf16(
        false, afrag, false, bfrag, (short)0, acc, false, false);
  }

  float bb = bias[colG];
#pragma unroll
  for (int v = 0; v < 8; ++v) {
    int rowG = mBase + half * 8 + v;
    float t = acc[v] + bb;
    if (doRelu) t = fmaxf(t, 0.f);
    if (outBf) outBf[(size_t)rowG * N + colG] = (bf16_t)t;
    if (outF)  outF[(size_t)rowG * N + colG]  = t;
  }
}

// ---------------------------------------------------------------------------
// Launch
// ---------------------------------------------------------------------------
extern "C" void kernel_launch(void* const* d_in, const int* in_sizes, int n_in,
                              void* d_out, int out_size, void* d_ws, size_t ws_size,
                              hipStream_t stream) {
  const float* xs0  = (const float*)d_in[0];
  const float* xs1  = (const float*)d_in[1];
  const float* xs2  = (const float*)d_in[2];
  const float* L0   = (const float*)d_in[3];
  const float* L1   = (const float*)d_in[4];
  const float* L2   = (const float*)d_in[5];
  const float* Wc0  = (const float*)d_in[6];   // [2, 4, 2048]
  const float* Wc1  = (const float*)d_in[7];   // [2, 4, 4096]
  const float* Wc2  = (const float*)d_in[8];   // [2, 4, 2048]
  const float* W_in = (const float*)d_in[9];   // [8192, 1024]
  const float* b_in = (const float*)d_in[10];
  const float* W_h  = (const float*)d_in[11];  // [1024, 1024]
  const float* b_h  = (const float*)d_in[12];
  const float* W_out= (const float*)d_in[13];  // [1024, 256]
  const float* b_out= (const float*)d_in[14];

  const int N0 = 2048, N1 = 4096, N2 = 2048;

  // ---- workspace carving (all offsets 256B-aligned by construction)
  char* ws = (char*)d_ws;
  size_t off = 0;
  bf16_t* Xping = (bf16_t*)(ws + off); off += (size_t)BS_DIM * IN_DIM * 2;   // 4 MiB
  bf16_t* Xpong = (bf16_t*)(ws + off); off += (size_t)BS_DIM * IN_DIM * 2;   // 4 MiB
  float*  Hcat  = (float*) (ws + off); off += (size_t)BS_DIM * IN_DIM * 4;   // 8 MiB
  bf16_t* Hb    = (bf16_t*)(ws + off); off += (size_t)B_DIM * IN_DIM * 2;    // 512 KiB
  bf16_t* h1    = (bf16_t*)(ws + off); off += (size_t)B_DIM * H_DIM * 2;
  bf16_t* h2    = (bf16_t*)(ws + off); off += (size_t)B_DIM * H_DIM * 2;
  bf16_t* WinT  = (bf16_t*)(ws + off); off += (size_t)H_DIM * IN_DIM * 2;    // 16 MiB
  bf16_t* WhT   = (bf16_t*)(ws + off); off += (size_t)H_DIM * H_DIM * 2;     // 2 MiB
  bf16_t* WoT   = (bf16_t*)(ws + off); off += (size_t)OUT_DIM * H_DIM * 2;   // 512 KiB
  (void)ws_size; (void)in_sizes; (void)n_in; (void)out_size;

  // ---- 1) convert inputs x -> bf16 into Xping (per-stack column bands)
  {
    int t0 = BS_DIM * N0, t1 = BS_DIM * N1, t2 = BS_DIM * N2;
    cvt_x_kernel<<<(t0 + 255) / 256, 256, 0, stream>>>(xs0, Xping, N0, 0, t0);
    cvt_x_kernel<<<(t1 + 255) / 256, 256, 0, stream>>>(xs1, Xping, N1, N0, t1);
    cvt_x_kernel<<<(t2 + 255) / 256, 256, 0, stream>>>(xs2, Xping, N2, N0 + N1, t2);
  }
  // ---- 2) transpose-convert MLP weights
  {
    int tw = IN_DIM * H_DIM;
    transpose_cvt_kernel<<<(tw + 255) / 256, 256, 0, stream>>>(W_in, WinT, IN_DIM, H_DIM, tw);
    int th = H_DIM * H_DIM;
    transpose_cvt_kernel<<<(th + 255) / 256, 256, 0, stream>>>(W_h, WhT, H_DIM, H_DIM, th);
    int to = H_DIM * OUT_DIM;
    transpose_cvt_kernel<<<(to + 255) / 256, 256, 0, stream>>>(W_out, WoT, H_DIM, OUT_DIM, to);
  }

  // ---- 3) conv layer 1 (per stack): Xping -> Xpong (bf16)
  conv_gemm_kernel<<<dim3(N0 / 16, 2), 32, 0, stream>>>(Xping, L0, Wc0, Xpong, nullptr, N0, 0);
  conv_gemm_kernel<<<dim3(N1 / 16, 2), 32, 0, stream>>>(Xping, L1, Wc1, Xpong, nullptr, N1, N0);
  conv_gemm_kernel<<<dim3(N2 / 16, 2), 32, 0, stream>>>(Xping, L2, Wc2, Xpong, nullptr, N2, N0 + N1);

  // ---- 4) conv layer 2 (per stack): Xpong -> Hcat (f32, full precision into MLP)
  conv_gemm_kernel<<<dim3(N0 / 16, 2), 32, 0, stream>>>(Xpong, L0, Wc0 + 1 * NFILT * N0, nullptr, Hcat, N0, 0);
  conv_gemm_kernel<<<dim3(N1 / 16, 2), 32, 0, stream>>>(Xpong, L1, Wc1 + 1 * NFILT * N1, nullptr, Hcat, N1, N0);
  conv_gemm_kernel<<<dim3(N2 / 16, 2), 32, 0, stream>>>(Xpong, L2, Wc2 + 1 * NFILT * N2, nullptr, Hcat, N2, N0 + N1);

  // ---- 5) sum over sequence -> Hb bf16 [32, 8192]
  {
    int ts = B_DIM * IN_DIM;
    seqsum_kernel<<<(ts + 255) / 256, 256, 0, stream>>>(Hcat, Hb, ts);
  }

  // ---- 6) MLP: h1 = relu(Hb @ W_in + b_in); h2 = relu(h1 @ W_h + b_h); out = h2 @ W_out + b_out
  mlp_gemm_kernel<<<dim3(H_DIM / 16, B_DIM / 16), 32, 0, stream>>>(Hb, WinT, b_in, h1, nullptr, IN_DIM, H_DIM, 1);
  mlp_gemm_kernel<<<dim3(H_DIM / 16, B_DIM / 16), 32, 0, stream>>>(h1, WhT, b_h, h2, nullptr, H_DIM, H_DIM, 1);
  mlp_gemm_kernel<<<dim3(OUT_DIM / 16, B_DIM / 16), 32, 0, stream>>>(h2, WoT, b_out, nullptr, (float*)d_out, H_DIM, OUT_DIM, 0);
}
